// ImageToGraphTransformer_60601988546923
// MI455X (gfx1250) — matrix operand order repaired and checked by reading, not compiled, not run on hardware
//
#include <hip/hip_runtime.h>
#include <hip/hip_fp16.h>

typedef __attribute__((ext_vector_type(16))) _Float16 v16h;
typedef __attribute__((ext_vector_type(8)))  float    v8f;

#define NPIX  65536
#define CIN   64
#define NHEAD 8
#define NROW  24      // (head, j) rows: j in {Wp.x, Wp.y, bp}
#define EMB   256
#define CHUNK 512     // pixels per accum block
#define NCHUNK 128    // NPIX / CHUNK

// ---------------------------------------------------------------- K0: prep
// q = z@Wq.T + bq ; U[b][3h+j][c] = sum_d q[b,h,d]*wsel(d,j)*Wk[h*32+d, c]
// svec[b][3h+j]   = sum_d q[b,h,d]*wsel(d,j)*bk[h*32+d]
__global__ __launch_bounds__(256)
void prep_kernel(const float* __restrict__ z,  const float* __restrict__ Wq,
                 const float* __restrict__ bq, const float* __restrict__ Wk,
                 const float* __restrict__ bk, const float* __restrict__ Wp,
                 const float* __restrict__ bp,
                 float* __restrict__ U, float* __restrict__ svec) {
  __shared__ float qsh[8 * EMB];
  const int t = threadIdx.x;            // 256 threads, t == e
  for (int b = 0; b < 8; ++b) {
    float acc = bq[t];
    for (int c = 0; c < CIN; ++c) acc += z[b * CIN + c] * Wq[t * CIN + c];
    qsh[b * EMB + t] = acc;
  }
  __syncthreads();
  for (int idx = t; idx < 8 * NROW * CIN; idx += 256) {
    int b  = idx / (NROW * CIN);
    int r  = idx % (NROW * CIN);
    int hj = r / CIN, c = r % CIN;
    int h = hj / 3, j = hj % 3;
    float acc = 0.f;
    for (int d = 0; d < 32; ++d) {
      float w = (j == 0) ? Wp[2 * d] : (j == 1) ? Wp[2 * d + 1] : bp[d];
      acc += qsh[b * EMB + h * 32 + d] * w * Wk[(h * 32 + d) * CIN + c];
    }
    U[idx] = acc;
  }
  if (t < 8 * NROW) {
    int b = t / NROW, hj = t % NROW;
    int h = hj / 3, j = hj % 3;
    float acc = 0.f;
    for (int d = 0; d < 32; ++d) {
      float w = (j == 0) ? Wp[2 * d] : (j == 1) ? Wp[2 * d + 1] : bp[d];
      acc += qsh[b * EMB + h * 32 + d] * w * bk[h * 32 + d];
    }
    svec[t] = acc;
  }
}

// ---------------------------------------------------------------- K1: logits (WMMA)
// D[24 x Ntile] = U (24x64, f16) @ X (64 x Ntile, f16), f32 accumulate.
// Two 16-row M-tiles, two K=32 steps. Per wave: 8 tiles of 16 pixels.
__global__ __launch_bounds__(256)
void logits_kernel(const float* __restrict__ x, const float* __restrict__ pos,
                   const float* __restrict__ U, const float* __restrict__ svec,
                   float* __restrict__ logits) {
  __shared__ float Ush[NROW * CIN];       // 6 KB
  __shared__ float Dsh[8 * 32 * 17];      // 17 KB, per-wave transpose scratch
  const int b    = blockIdx.y;
  const int t    = threadIdx.x;
  const int wave = t >> 5;
  const int lane = t & 31;
  const int col  = lane & 15;
  const bool hi  = lane >= 16;

  for (int i = t; i < NROW * CIN; i += 256) Ush[i] = U[b * NROW * CIN + i];
  __syncthreads();

  // A fragments, contiguous K order matching B:
  // lanes 0-15: M=lane,  K in {0..15}; lanes 16-31: M=lane-16, K in {16..31}
  v16h A[2][2];
#pragma unroll
  for (int mt = 0; mt < 2; ++mt)
#pragma unroll
    for (int ks = 0; ks < 2; ++ks) {
      const int M   = mt * 16 + col;
      const int klo = hi ? 16 : 0;
#pragma unroll
      for (int i = 0; i < 16; ++i) {
        const int K = klo + i;
        const float v = (M < NROW) ? Ush[M * CIN + ks * 32 + K] : 0.f;
        A[mt][ks][i] = (_Float16)v;
      }
    }

  const int hbase = hi ? 4 : 0;
  float sa[4], sb[4], sc[4], px[4], py[4];
#pragma unroll
  for (int hh = 0; hh < 4; ++hh) {
    const int h = hbase + hh;
    sa[hh] = svec[b * NROW + 3 * h + 0];
    sb[hh] = svec[b * NROW + 3 * h + 1];
    sc[hh] = svec[b * NROW + 3 * h + 2];
    px[hh] = pos[(b * NHEAD + h) * 2 + 0];
    py[hh] = pos[(b * NHEAD + h) * 2 + 1];
  }

  float* scratch = &Dsh[wave * 32 * 17];
  const float* xb = x + (size_t)b * CIN * NPIX;
  const float invs = 0.17677669529663687f;   // 1/sqrt(32)
  const int n_block0 = blockIdx.x * 1024;

  for (int tile = 0; tile < 8; ++tile) {
    const int n0  = n_block0 + wave * 128 + tile * 16;
    const int pix = n0 + col;
    // B in contiguous K order matching A: lanes 0-15 hold K=0..15, lanes 16-31 K=16..31
    v16h B0, B1;
    const int kb = hi ? 16 : 0;
#pragma unroll
    for (int i = 0; i < 16; ++i) {
      B0[i] = (_Float16)xb[(size_t)(kb + i) * NPIX + pix];
      B1[i] = (_Float16)xb[(size_t)(32 + kb + i) * NPIX + pix];
    }
    v8f acc0 = {}; v8f acc1 = {};
    acc0 = __builtin_amdgcn_wmma_f32_16x16x32_f16(false, A[0][0], false, B0, (short)0, acc0, false, false);
    acc0 = __builtin_amdgcn_wmma_f32_16x16x32_f16(false, A[0][1], false, B1, (short)0, acc0, false, false);
    acc1 = __builtin_amdgcn_wmma_f32_16x16x32_f16(false, A[1][0], false, B0, (short)0, acc1, false, false);
    acc1 = __builtin_amdgcn_wmma_f32_16x16x32_f16(false, A[1][1], false, B1, (short)0, acc1, false, false);

    // D layout: VGPR r holds M = r + (hi?8:0), N = col
    const int rb = hi ? 8 : 0;
#pragma unroll
    for (int r = 0; r < 8; ++r) {
      scratch[(rb + r) * 17 + col]      = acc0[r];
      scratch[(16 + rb + r) * 17 + col] = acc1[r];
    }
    asm volatile("s_wait_dscnt 0x0" ::: "memory");

    const int n = n0 + col;
    const float xx = (float)(n >> 8);
    const float yy = (float)(n & 255);
    float out[4];
#pragma unroll
    for (int hh = 0; hh < 4; ++hh) {
      const int h = hbase + hh;
      const float dA = scratch[(3 * h + 0) * 17 + col];
      const float dB = scratch[(3 * h + 1) * 17 + col];
      const float dC = scratch[(3 * h + 2) * 17 + col];
      out[hh] = ((xx - px[hh]) * (dA + sa[hh]) +
                 (yy - py[hh]) * (dB + sb[hh]) + (dC + sc[hh])) * invs;
    }
    *(float4*)(logits + ((size_t)b * NPIX + n) * 8 + hbase) =
        make_float4(out[0], out[1], out[2], out[3]);
    asm volatile("s_wait_dscnt 0x0" ::: "memory");
  }
}

// ---------------------------------------------------------------- K2: per-(b,h) max
__global__ __launch_bounds__(256)
void max_kernel(const float* __restrict__ logits, float* __restrict__ maxv) {
  __shared__ float red[256];
  const int b = blockIdx.x, t = threadIdx.x;
  const float* lg = logits + (size_t)b * NPIX * 8;
  float m = -3.4e38f;                      // h = (t + k*256) % 8 == t % 8: fixed per thread
  for (int i = t; i < NPIX * 8; i += 256) m = fmaxf(m, lg[i]);
  red[t] = m;
  __syncthreads();
  if (t < 8) {
    float mm = red[t];
    for (int k = 1; k < 32; ++k) mm = fmaxf(mm, red[t + 8 * k]);
    maxv[b * 8 + t] = mm;
  }
}

// ---------------------------------------------------------------- K3: weighted partial sums
__global__ __launch_bounds__(512)
void accum_kernel(const float* __restrict__ x, const float* __restrict__ logits,
                  const float* __restrict__ maxv, float* __restrict__ partials) {
  __shared__ float wsh[CHUNK * 8];         // 16 KB
  const int b = blockIdx.y, chunk = blockIdx.x, t = threadIdx.x;   // 512 threads
  const int n0 = chunk * CHUNK;
  {
    const int n = n0 + t;
    const float4* lp = (const float4*)(logits + ((size_t)b * NPIX + n) * 8);
    const float4 l0 = lp[0], l1 = lp[1];
    const float k = 1.4426950408889634f;
    wsh[t * 8 + 0] = exp2f((l0.x - maxv[b * 8 + 0]) * k);
    wsh[t * 8 + 1] = exp2f((l0.y - maxv[b * 8 + 1]) * k);
    wsh[t * 8 + 2] = exp2f((l0.z - maxv[b * 8 + 2]) * k);
    wsh[t * 8 + 3] = exp2f((l0.w - maxv[b * 8 + 3]) * k);
    wsh[t * 8 + 4] = exp2f((l1.x - maxv[b * 8 + 4]) * k);
    wsh[t * 8 + 5] = exp2f((l1.y - maxv[b * 8 + 5]) * k);
    wsh[t * 8 + 6] = exp2f((l1.z - maxv[b * 8 + 6]) * k);
    wsh[t * 8 + 7] = exp2f((l1.w - maxv[b * 8 + 7]) * k);
  }
  __syncthreads();
  const int h = t >> 6, c = t & 63;
  const float* xr = x + (size_t)b * CIN * NPIX + (size_t)c * NPIX + n0;
  float S = 0.f;
  for (int p = 0; p < CHUNK; ++p) S += wsh[p * 8 + h] * xr[p];
  float* out = partials + (((size_t)(b * NCHUNK + chunk)) * 8 + h) * 68;
  out[c] = S;
  if (c == 0) {
    float Z = 0.f, Px = 0.f, Py = 0.f;
    for (int p = 0; p < CHUNK; ++p) {
      const float w = wsh[p * 8 + h];
      const int n = n0 + p;
      Z += w; Px += w * (float)(n >> 8); Py += w * (float)(n & 255);
    }
    out[64] = Z; out[65] = Px; out[66] = Py;
  }
}

// ---------------------------------------------------------------- K4: reduce + normalize + Wv GEMM
__global__ __launch_bounds__(256)
void finalize_kernel(const float* __restrict__ partials, const float* __restrict__ Wv,
                     const float* __restrict__ bv, float* __restrict__ out) {
  __shared__ float tot[68];
  const int blk = blockIdx.x, b = blk >> 3, h = blk & 7;
  const int t = threadIdx.x;               // 256 threads, t == e
  if (t < 68) {
    float acc = 0.f;
    for (int ch = 0; ch < NCHUNK; ++ch)
      acc += partials[(((size_t)(b * NCHUNK + ch)) * 8 + h) * 68 + t];
    tot[t] = acc;
  }
  __syncthreads();
  const float invZ = 1.0f / tot[64];
  float dot = 0.f;
  for (int c = 0; c < CIN; ++c) dot += Wv[t * CIN + c] * tot[c];
  out[(b * 8 + h) * EMB + t] = bv[t] + invZ * dot;
  if (t == 0) {
    out[8 * 8 * EMB + (b * 8 + h) * 2 + 0] = tot[65] * invZ;
    out[8 * 8 * EMB + (b * 8 + h) * 2 + 1] = tot[66] * invZ;
  }
}

// ---------------------------------------------------------------- launch
extern "C" void kernel_launch(void* const* d_in, const int* in_sizes, int n_in,
                              void* d_out, int out_size, void* d_ws, size_t ws_size,
                              hipStream_t stream) {
  const float* x   = (const float*)d_in[0];
  const float* z   = (const float*)d_in[1];
  const float* pos = (const float*)d_in[2];
  const float* Wq  = (const float*)d_in[3];
  const float* bq  = (const float*)d_in[4];
  const float* Wk  = (const float*)d_in[5];
  const float* bk  = (const float*)d_in[6];
  const float* Wv  = (const float*)d_in[7];
  const float* bv  = (const float*)d_in[8];
  const float* Wp  = (const float*)d_in[9];
  const float* bp  = (const float*)d_in[10];
  float* out = (float*)d_out;
  float* ws  = (float*)d_ws;

  float* U        = ws;                       // 8*24*64   = 12288 floats
  float* svec     = ws + 12288;               // 192
  float* maxv     = ws + 12480;               // 64
  float* logits   = ws + 16384;               // 8*65536*8 = 4194304 floats (16 MB)
  float* partials = ws + 16384 + 4194304;     // 8*128*8*68 = 557056 floats (~2.2 MB)

  prep_kernel<<<1, 256, 0, stream>>>(z, Wq, bq, Wk, bk, Wp, bp, U, svec);
  logits_kernel<<<dim3(64, 8), 256, 0, stream>>>(x, pos, U, svec, logits);
  max_kernel<<<8, 256, 0, stream>>>(logits, maxv);
  accum_kernel<<<dim3(NCHUNK, 8), 512, 0, stream>>>(x, logits, maxv, partials);
  finalize_kernel<<<64, 256, 0, stream>>>(partials, Wv, bv, out);
}
